// LSTM_28561532518805
// MI455X (gfx1250) — matrix operand order repaired and checked by reading, not compiled, not run on hardware
//
#include <hip/hip_runtime.h>
#include <hip/hip_bf16.h>

// LSTM on MI455X (gfx1250):
//  Phase 0: transpose+cvt weights to bf16 [N][K]; convert x to bf16 (streaming pass) so the
//           GEMM hot loops contain only b128 loads + v_wmma (no per-element cvt VALU).
//  Phase 1: bf16 WMMA GEMM  xproj = x @ [Wxi|Wxf|Wxc|Wxo] + bx   (32768x1024 * 1024x4096)
//           wave tile 32x64 (2 A-frags reused over 4 B-frags -> 8 WMMA per k-step).
//  Phase 2: persistent kernel, 64 blocks (16 h-columns each), 512 steps with an atomic grid
//           barrier per step; 4 gate tiles + y tile fused on shared A fragments.
// Workspace required: ~620 MB.

#define S_LEN 512
#define BATCH 64
#define HDIM  1024
#define GDIM  4096  // 4*HDIM (i,f,c,o concatenated)

typedef __attribute__((ext_vector_type(16))) __bf16   v16bf;
typedef __attribute__((ext_vector_type(8)))  float    v8f;
typedef __attribute__((ext_vector_type(8)))  unsigned v8u;

__device__ __forceinline__ unsigned short f2bf(float f) {
  unsigned u = __float_as_uint(f);
  u += 0x7FFFu + ((u >> 16) & 1u);          // round-to-nearest-even
  return (unsigned short)(u >> 16);
}
__device__ __forceinline__ unsigned pack2(float lo, float hi) {
  return (unsigned)f2bf(lo) | ((unsigned)f2bf(hi) << 16);
}
__device__ __forceinline__ float fsigmoid(float x) { return 1.0f / (1.0f + __expf(-x)); }
__device__ __forceinline__ float ftanh(float x)    { return 2.0f / (1.0f + __expf(-2.0f * x)) - 1.0f; }

// B fragment: lane n = column (lanes 0-15) / same columns K+16 (lanes 16-31).
// Weights stored transposed [N][K] bf16 -> 16 contiguous bf16 per lane (2x b128).
__device__ __forceinline__ v16bf load_bfrag(const unsigned short* p) {
  uint4 r0 = *(const uint4*)(p);
  uint4 r1 = *(const uint4*)(p + 8);
  v8u u = {r0.x, r0.y, r0.z, r0.w, r1.x, r1.y, r1.z, r1.w};
  return __builtin_bit_cast(v16bf, u);
}
// A fragment (bf16 row-major [M][K]): lane holds K kb..kb+15 contiguous.
__device__ __forceinline__ v16bf load_afrag_bf(const unsigned short* p) {
  uint4 q0 = *(const uint4*)(p);
  uint4 q1 = *(const uint4*)(p + 8);
  v8u u = {q0.x, q0.y, q0.z, q0.w, q1.x, q1.y, q1.z, q1.w};
  return __builtin_bit_cast(v16bf, u);
}
#define WMMA_BF16(a, b, c) \
  __builtin_amdgcn_wmma_f32_16x16x32_bf16(false, (a), false, (b), (short)0, (c), false, false)

// ---------------- Phase 0a: tiled transpose + fp32->bf16 convert (weights) ----------------
__global__ __launch_bounds__(256) void k_transpose_cvt(const float* __restrict__ src,
                                                       unsigned short* __restrict__ dst,
                                                       int rows, int cols) {
  __shared__ float tile[32][33];
  const int tx = threadIdx.x & 31;
  const int ty = threadIdx.x >> 5;  // 0..7
  const int c0 = blockIdx.x * 32;
  const int r0 = blockIdx.y * 32;
#pragma unroll
  for (int i = ty; i < 32; i += 8)
    tile[i][tx] = src[(size_t)(r0 + i) * cols + (c0 + tx)];
  __syncthreads();
#pragma unroll
  for (int i = ty; i < 32; i += 8)
    dst[(size_t)(c0 + i) * rows + (r0 + tx)] = f2bf(tile[tx][i]);
}

// ---------------- Phase 0b: streaming fp32 -> bf16 (x), bandwidth-bound ----------------
__global__ __launch_bounds__(256) void k_cvt_bf16(const float* __restrict__ src,
                                                  unsigned short* __restrict__ dst, size_t n4) {
  size_t i = (size_t)blockIdx.x * blockDim.x + threadIdx.x;
  if (i < n4) {
    float4 f = ((const float4*)src)[i];
    uint2 o; o.x = pack2(f.x, f.y); o.y = pack2(f.z, f.w);
    ((uint2*)dst)[i] = o;
  }
}

__global__ void k_zero(unsigned* __restrict__ p, size_t n) {
  size_t i = (size_t)blockIdx.x * blockDim.x + threadIdx.x;
  if (i < n) p[i] = 0u;
}

// ---------------- Phase 1: xproj = xbf @ WxT^T + bx  (bf16 WMMA, f32 acc) ----------------
// Block tile 64(M) x 256(N); 8 waves, wave tile 32x64 (2x4 WMMA tiles).
__global__ __launch_bounds__(256) void k_xproj(const unsigned short* __restrict__ xbf, // [32768][1024] bf16
                                               const unsigned short* __restrict__ WxT, // [4096][1024] bf16
                                               const float* __restrict__ bxi, const float* __restrict__ bxf,
                                               const float* __restrict__ bxc, const float* __restrict__ bxo,
                                               float* __restrict__ xproj) {            // [32768][4096]
  const int lane = threadIdx.x & 31;
  const int wid  = threadIdx.x >> 5;     // 0..7
  const int lm = lane & 15, lh = lane >> 4;
  const int M0 = blockIdx.y * 64 + (wid >> 2) * 32;   // 2 M-waves
  const int N0 = blockIdx.x * 256 + (wid & 3) * 64;   // 4 N-waves

  const unsigned short* ap0 = xbf + (size_t)(M0 + lm) * HDIM + lh * 16;
  const unsigned short* ap1 = xbf + (size_t)(M0 + 16 + lm) * HDIM + lh * 16;
  const unsigned short* bp0 = WxT + (size_t)(N0 + 0 * 16 + lm) * HDIM + lh * 16;
  const unsigned short* bp1 = WxT + (size_t)(N0 + 1 * 16 + lm) * HDIM + lh * 16;
  const unsigned short* bp2 = WxT + (size_t)(N0 + 2 * 16 + lm) * HDIM + lh * 16;
  const unsigned short* bp3 = WxT + (size_t)(N0 + 3 * 16 + lm) * HDIM + lh * 16;

  v8f acc[2][4] = {};
#pragma unroll 2
  for (int kt = 0; kt < HDIM; kt += 32) {
    v16bf a0 = load_afrag_bf(ap0 + kt);
    v16bf a1 = load_afrag_bf(ap1 + kt);
    v16bf b;
    b = load_bfrag(bp0 + kt);
    acc[0][0] = WMMA_BF16(a0, b, acc[0][0]);
    acc[1][0] = WMMA_BF16(a1, b, acc[1][0]);
    b = load_bfrag(bp1 + kt);
    acc[0][1] = WMMA_BF16(a0, b, acc[0][1]);
    acc[1][1] = WMMA_BF16(a1, b, acc[1][1]);
    b = load_bfrag(bp2 + kt);
    acc[0][2] = WMMA_BF16(a0, b, acc[0][2]);
    acc[1][2] = WMMA_BF16(a1, b, acc[1][2]);
    b = load_bfrag(bp3 + kt);
    acc[0][3] = WMMA_BF16(a0, b, acc[0][3]);
    acc[1][3] = WMMA_BF16(a1, b, acc[1][3]);
  }

  const float* bx[4] = {bxi, bxf, bxc, bxo};
#pragma unroll
  for (int mt = 0; mt < 2; ++mt)
#pragma unroll
    for (int nt = 0; nt < 4; ++nt) {
      const int ncol0 = N0 + nt * 16;
      const int col = ncol0 + lm;
      const float bias = bx[ncol0 >> 10][col & (HDIM - 1)];
      const int row = M0 + mt * 16 + lh * 8;
      float* o = xproj + (size_t)row * GDIM + col;
      v8f a = acc[mt][nt];
#pragma unroll
      for (int r = 0; r < 8; ++r)
        o[(size_t)r * GDIM] = a[r] + bias;
    }
}

// ---------------- Phase 2: persistent recurrent kernel ----------------
// 64 blocks x 128 threads (4 waves). Block b owns h-columns [16b,16b+16).
// Iteration t: GEMM A = h_t (bf16, double-buffered). Computes:
//   gates pre-act (4 N-tiles) -> h_{t+1}, c_{t+1}   (t < S)
//   y_{t-1} = h_t @ W_hy      (1 N-tile)            (t >= 1)
// One grid-wide atomic barrier per iteration.
__global__ __launch_bounds__(128) void k_lstm_recurrent(
    const unsigned short* __restrict__ WhT,  // [4096][1024] bf16
    const unsigned short* __restrict__ WyT,  // [1024][1024] bf16
    const float* __restrict__ xproj,         // [S*64][4096]
    const float* __restrict__ bhi, const float* __restrict__ bhf,
    const float* __restrict__ bhc, const float* __restrict__ bho,
    const float* __restrict__ bhy,
    unsigned short* __restrict__ h0buf, unsigned short* __restrict__ h1buf,
    float* __restrict__ cbuf, unsigned* __restrict__ bar,
    float* __restrict__ out) {
  const int lane = threadIdx.x & 31;
  const int w    = threadIdx.x >> 5;   // M strip (16 batch rows each)
  const int lm = lane & 15, lh = lane >> 4;
  const int hcol0 = blockIdx.x * 16;
  const int col = hcol0 + lm;

  const float bh0 = bhi[col], bh1 = bhf[col], bh2 = bhc[col], bh3 = bho[col], bhyl = bhy[col];

  const unsigned short* bp0 = WhT + ((size_t)(0 * HDIM + hcol0 + lm)) * HDIM + lh * 16;
  const unsigned short* bp1 = WhT + ((size_t)(1 * HDIM + hcol0 + lm)) * HDIM + lh * 16;
  const unsigned short* bp2 = WhT + ((size_t)(2 * HDIM + hcol0 + lm)) * HDIM + lh * 16;
  const unsigned short* bp3 = WhT + ((size_t)(3 * HDIM + hcol0 + lm)) * HDIM + lh * 16;
  const unsigned short* bp4 = WyT + ((size_t)(hcol0 + lm)) * HDIM + lh * 16;

  unsigned short* hbufs[2] = {h0buf, h1buf};
  const size_t arow = (size_t)(w * 16 + lm) * HDIM + lh * 16;

  for (int t = 0; t <= S_LEN; ++t) {
    const bool do_gates = (t < S_LEN);
    v8f acc0 = {}, acc1 = {}, acc2 = {}, acc3 = {}, acc4 = {};

    if (t > 0) {
      const unsigned short* ap = hbufs[t & 1] + arow;
#pragma unroll 2
      for (int kt = 0; kt < HDIM; kt += 32) {
        v16bf a = load_afrag_bf(ap + kt);
        if (do_gates) {
          acc0 = WMMA_BF16(a, load_bfrag(bp0 + kt), acc0);
          acc1 = WMMA_BF16(a, load_bfrag(bp1 + kt), acc1);
          acc2 = WMMA_BF16(a, load_bfrag(bp2 + kt), acc2);
          acc3 = WMMA_BF16(a, load_bfrag(bp3 + kt), acc3);
        }
        acc4 = WMMA_BF16(a, load_bfrag(bp4 + kt), acc4);  // y_{t-1}
      }
    }

    const int row = w * 16 + lh * 8;
    if (do_gates) {
      const float* xp = xproj + ((size_t)t * BATCH + row) * GDIM + col;
      float* cp = cbuf + (size_t)row * HDIM + col;
      unsigned short* hp = hbufs[(t + 1) & 1] + (size_t)row * HDIM + col;
#pragma unroll
      for (int r = 0; r < 8; ++r) {
        const float* xr = xp + (size_t)r * GDIM;
        float ig = fsigmoid(acc0[r] + xr[0]        + bh0);
        float fg = fsigmoid(acc1[r] + xr[HDIM]     + bh1);
        float gg = ftanh   (acc2[r] + xr[2 * HDIM] + bh2);
        float og = fsigmoid(acc3[r] + xr[3 * HDIM] + bh3);
        float cv = cp[(size_t)r * HDIM];
        float cn = fg * cv + ig * gg;
        float hn = og * ftanh(cn);
        cp[(size_t)r * HDIM] = cn;
        hp[(size_t)r * HDIM] = f2bf(hn);
      }
    }
    if (t >= 1) {
      float* op = out + ((size_t)(t - 1) * BATCH + row) * HDIM + col;
#pragma unroll
      for (int r = 0; r < 8; ++r)
        op[(size_t)r * HDIM] = acc4[r] + bhyl;
    }

    // grid-wide barrier (all 64 blocks resident)
    __threadfence();
    __syncthreads();
    if (threadIdx.x == 0) {
      __hip_atomic_fetch_add(bar, 1u, __ATOMIC_ACQ_REL, __HIP_MEMORY_SCOPE_AGENT);
      const unsigned target = (unsigned)(t + 1) * gridDim.x;
      while (__hip_atomic_load(bar, __ATOMIC_ACQUIRE, __HIP_MEMORY_SCOPE_AGENT) < target)
        __builtin_amdgcn_s_sleep(1);
    }
    __syncthreads();
  }
}

extern "C" void kernel_launch(void* const* d_in, const int* in_sizes, int n_in,
                              void* d_out, int out_size, void* d_ws, size_t ws_size,
                              hipStream_t stream) {
  (void)in_sizes; (void)n_in; (void)out_size; (void)ws_size;
  const float* x     = (const float*)d_in[0];
  const float* Wx[4] = {(const float*)d_in[1], (const float*)d_in[3],
                        (const float*)d_in[5], (const float*)d_in[7]};
  const float* bx[4] = {(const float*)d_in[2], (const float*)d_in[4],
                        (const float*)d_in[6], (const float*)d_in[8]};
  const float* Wh[4] = {(const float*)d_in[9],  (const float*)d_in[11],
                        (const float*)d_in[13], (const float*)d_in[15]};
  const float* bh[4] = {(const float*)d_in[10], (const float*)d_in[12],
                        (const float*)d_in[14], (const float*)d_in[16]};
  const float* Why = (const float*)d_in[17];
  const float* bhy = (const float*)d_in[18];

  char* ws = (char*)d_ws;
  size_t off = 0;
  unsigned short* WxT = (unsigned short*)(ws + off); off += (size_t)GDIM * HDIM * 2;   // 8 MB
  unsigned short* WhT = (unsigned short*)(ws + off); off += (size_t)GDIM * HDIM * 2;   // 8 MB
  unsigned short* WyT = (unsigned short*)(ws + off); off += (size_t)HDIM * HDIM * 2;   // 2 MB
  unsigned short* h0  = (unsigned short*)(ws + off); off += (size_t)BATCH * HDIM * 2;
  unsigned short* h1  = (unsigned short*)(ws + off); off += (size_t)BATCH * HDIM * 2;
  float*    cbuf = (float*)(ws + off);    off += (size_t)BATCH * HDIM * 4;
  unsigned* bar  = (unsigned*)(ws + off); off += 256;
  unsigned short* xbf = (unsigned short*)(ws + off); off += (size_t)S_LEN * BATCH * HDIM * 2; // 64 MB
  float*    xproj = (float*)(ws + off);   off += (size_t)S_LEN * BATCH * GDIM * 4;            // 512 MB

  // Phase 0a: weight transposes (fp32 -> bf16 [N][K])
  dim3 tb(256), tg(HDIM / 32, HDIM / 32);
  for (int g = 0; g < 4; ++g) {
    k_transpose_cvt<<<tg, tb, 0, stream>>>(Wx[g], WxT + (size_t)g * HDIM * HDIM, HDIM, HDIM);
    k_transpose_cvt<<<tg, tb, 0, stream>>>(Wh[g], WhT + (size_t)g * HDIM * HDIM, HDIM, HDIM);
  }
  k_transpose_cvt<<<tg, tb, 0, stream>>>(Why, WyT, HDIM, HDIM);

  // Phase 0b: x -> bf16 (streaming)
  size_t n4 = (size_t)S_LEN * BATCH * HDIM / 4;
  k_cvt_bf16<<<(unsigned)((n4 + 255) / 256), 256, 0, stream>>>(x, xbf, n4);

  // zero h/c state and barrier counter (contiguous region starting at h0)
  size_t zero_words = ((size_t)BATCH * HDIM * 2 * 2 + (size_t)BATCH * HDIM * 4 + 256) / 4;
  k_zero<<<(unsigned)((zero_words + 255) / 256), 256, 0, stream>>>((unsigned*)h0, zero_words);

  // Phase 1: input projections (block tile 64x256, wave tile 32x64)
  k_xproj<<<dim3(GDIM / 256, (S_LEN * BATCH) / 64), 256, 0, stream>>>(
      xbf, WxT, bx[0], bx[1], bx[2], bx[3], xproj);

  // Phase 2: persistent recurrent kernel
  k_lstm_recurrent<<<dim3(HDIM / 16), 128, 0, stream>>>(
      WhT, WyT, xproj, bh[0], bh[1], bh[2], bh[3], bhy, h0, h1, cbuf, bar, (float*)d_out);
}
